// RNN_Concat_33071248179756
// MI455X (gfx1250) — matrix operand order repaired and checked by reading, not compiled, not run on hardware
//
#include <hip/hip_runtime.h>

#define DEVFN __device__ __forceinline__

typedef __attribute__((ext_vector_type(16))) __bf16 v16bf;
typedef __attribute__((ext_vector_type(8)))  __bf16 v8bf;
typedef __attribute__((ext_vector_type(8)))  float  v8f;
typedef __attribute__((ext_vector_type(4)))  float  v4f;
typedef __attribute__((ext_vector_type(4)))  unsigned int v4u;
typedef __attribute__((ext_vector_type(4)))  int    v4i;

namespace {
constexpr int Bsz  = 2048;   // batch
constexpr int Hd   = 512;    // hidden
constexpr int Lv   = 64;     // sequence length
constexpr int FDim = 514;    // H + S (weight row stride)
constexpr int Kd   = 512;    // effective K (sigma cols handled in epilogue)
constexpr int MT   = 128;    // workgroup tile M (batch)
constexpr int NT   = 64;     // workgroup tile N (hidden out)
constexpr int KT   = 32;     // K tile == WMMA K
constexpr int NKT  = Kd / KT;
}

#if __has_builtin(__builtin_amdgcn_global_load_async_to_lds_b128)
#define HAS_ASYNC_LDS 1
#else
#define HAS_ASYNC_LDS 0
#endif

DEVFN void wait_async_lds() {
#if HAS_ASYNC_LDS
#if __has_builtin(__builtin_amdgcn_s_wait_asynccnt)
  __builtin_amdgcn_s_wait_asynccnt(0);
#else
  asm volatile("s_wait_asynccnt 0" ::: "memory");
#endif
#endif
}

DEVFN unsigned fbits(float x) { return __builtin_bit_cast(unsigned, x); }
DEVFN __bf16 to_bf16(float x) {
  unsigned short h = (unsigned short)(fbits(x) >> 16);
  return __builtin_bit_cast(__bf16, h);
}
DEVFN float from_bf16(__bf16 x) {
  unsigned u = ((unsigned)__builtin_bit_cast(unsigned short, x)) << 16;
  return __builtin_bit_cast(float, u);
}

// Y[b,n] = elu( sum_k X[b,k]*W[n,k] + bias[n] + (sig ? W[n, 512+sig[b]] : 0) )
// X: (Bsz x Kd) bf16 row-major, W: (Hd x FDim) fp32 row-major, Y: (Bsz x Hd) bf16
__global__ __launch_bounds__(256)
void fused_gemm_elu_bf16(const __bf16* __restrict__ X, const float* __restrict__ W,
                         const float* __restrict__ bias, const int* __restrict__ sig,
                         int sig_stride, __bf16* __restrict__ Y) {
  __shared__ __bf16 lX[2][MT * KT];
  __shared__ __bf16 lW[2][NT * KT];

  const int t    = threadIdx.x;
  const int lane = t & 31;
  const int wave = t >> 5;
  const int mBase = blockIdx.x * MT;
  const int nBase = blockIdx.y * NT;
  const int wM = (wave >> 1) * 32;   // wave tile offset in M (0..96)
  const int wN = (wave & 1) * 32;    // wave tile offset in N (0..32)
  const int lh = lane & 15;
  const int lq = lane >> 4;

  v8f acc[2][2] = {};

  auto stage = [&](int buf, int kBase) {
    // --- X tile: MT x KT bf16 = 8 KB; 2 x 16B chunks per thread, async -> LDS ---
#pragma unroll
    for (int r = 0; r < 2; ++r) {
      int c   = t + r * 256;      // chunk id 0..511
      int row = c >> 2;           // 4 x 16B chunks per 64B row
      int cc  = c & 3;
      const __bf16* g = X + (size_t)(mBase + row) * Kd + kBase + cc * 8;
      __bf16* l = &lX[buf][row * KT + cc * 8];
#if HAS_ASYNC_LDS
      __builtin_amdgcn_global_load_async_to_lds_b128(
          (__attribute__((address_space(1))) v4i*)g,
          (__attribute__((address_space(3))) v4i*)l, 0, 0);
#else
      *(v4u*)l = *(const v4u*)g;  // global_load_b128 + ds_store_b128 fallback
#endif
    }
    // --- W tile: NT x KT fp32 -> bf16 (on-the-fly convert), 8 floats/thread ---
    {
      int row = t >> 2;           // 0..63
      int cc  = t & 3;
      const float* g = W + (size_t)(nBase + row) * FDim + kBase + cc * 8;
      __builtin_prefetch(g + KT, 0, 1);   // global_prefetch_b8 on next K tile
      v4f a = *(const v4f*)g;
      v4f b = *(const v4f*)(g + 4);
      v4u p;
      p.x = (fbits(a.x) >> 16) | (fbits(a.y) & 0xffff0000u);
      p.y = (fbits(a.z) >> 16) | (fbits(a.w) & 0xffff0000u);
      p.z = (fbits(b.x) >> 16) | (fbits(b.y) & 0xffff0000u);
      p.w = (fbits(b.z) >> 16) | (fbits(b.w) & 0xffff0000u);
      *(v4u*)&lW[buf][row * KT + cc * 8] = p;
    }
  };

  auto compute = [&](int buf) {
    v16bf afrag[2], bfrag[2];
    // A frag 16x32 bf16: lane<16 holds row M=lane, K={0..7,16..23}; lane>=16: K={8..15,24..31}
#pragma unroll
    for (int i = 0; i < 2; ++i) {
      const __bf16* p = &lX[buf][(wM + i * 16 + lh) * KT + lq * 8];
      v8bf lo = *(const v8bf*)p;
      v8bf hi = *(const v8bf*)(p + 16);
      afrag[i] = __builtin_shufflevector(lo, hi, 0, 1, 2, 3, 4, 5, 6, 7,
                                                 8, 9, 10, 11, 12, 13, 14, 15);
    }
    // B frag 32x16 bf16: lanes0-15 col N=lane K={0..7,16..23}; lanes16-31 same cols K={8..15,24..31}
#pragma unroll
    for (int j = 0; j < 2; ++j) {
      const __bf16* p = &lW[buf][(wN + j * 16 + lh) * KT + lq * 8];
      v8bf lo = *(const v8bf*)p;
      v8bf hi = *(const v8bf*)(p + 16);
      bfrag[j] = __builtin_shufflevector(lo, hi, 0, 1, 2, 3, 4, 5, 6, 7,
                                                 8, 9, 10, 11, 12, 13, 14, 15);
    }
#pragma unroll
    for (int i = 0; i < 2; ++i)
#pragma unroll
      for (int j = 0; j < 2; ++j)
        acc[i][j] = __builtin_amdgcn_wmma_f32_16x16x32_bf16(
            false, afrag[i], false, bfrag[j], (short)0, acc[i][j], false, false);
  };

  int buf = 0;
  stage(0, 0);
  wait_async_lds();
  __syncthreads();
#pragma unroll 1
  for (int kt = 0; kt < NKT; ++kt) {
    if (kt + 1 < NKT) stage(buf ^ 1, (kt + 1) * KT);
    compute(buf);
    wait_async_lds();
    __syncthreads();
    buf ^= 1;
  }

  // Epilogue: C/D layout — VGPR r: lanes0-15 -> M=r,N=lane; lanes16-31 -> M=r+8,N=lane-16
#pragma unroll
  for (int i = 0; i < 2; ++i) {
#pragma unroll
    for (int j = 0; j < 2; ++j) {
#pragma unroll
      for (int r = 0; r < 8; ++r) {
        int m = mBase + wM + i * 16 + r + lq * 8;
        int n = nBase + wN + j * 16 + lh;
        float v = acc[i][j][r] + bias[n];
        if (sig)  // one-hot sigma contribution: column 512+sig of W (uniform branch)
          v += W[(size_t)n * FDim + Kd + sig[(size_t)m * sig_stride]];
        v = v > 0.f ? v : (__expf(v) - 1.f);  // ELU
        Y[(size_t)m * Hd + n] = to_bf16(v);
      }
    }
  }
}

// out[b, l] = softmax(h3[b]·Ws[l]^T + bs[l])[1] = sigmoid(z1 - z0); one wave per row b
__global__ __launch_bounds__(256)
void logits_sigmoid(const __bf16* __restrict__ H3, const float* __restrict__ Wsl,
                    const float* __restrict__ bsl, float* __restrict__ out, int l) {
  const int lane = threadIdx.x & 31;
  const int wave = threadIdx.x >> 5;
  const int b = blockIdx.x * 8 + wave;
  float z0 = 0.f, z1 = 0.f;
  for (int h = lane; h < Hd; h += 32) {
    float x = from_bf16(H3[(size_t)b * Hd + h]);
    z0 += x * Wsl[h];
    z1 += x * Wsl[Hd + h];
  }
#pragma unroll
  for (int off = 16; off > 0; off >>= 1) {
    z0 += __shfl_xor(z0, off, 32);
    z1 += __shfl_xor(z1, off, 32);
  }
  if (lane == 0) {
    float d = (z1 + bsl[1]) - (z0 + bsl[0]);
    out[(size_t)b * Lv + l] = 1.f / (1.f + __expf(-d));
  }
}

extern "C" void kernel_launch(void* const* d_in, const int* in_sizes, int n_in,
                              void* d_out, int out_size, void* d_ws, size_t ws_size,
                              hipStream_t stream) {
  (void)in_sizes; (void)n_in; (void)out_size; (void)ws_size;
  const int*   sigma_list = (const int*)d_in[0];
  const float* W1 = (const float*)d_in[1];
  const float* b1 = (const float*)d_in[2];
  const float* W2 = (const float*)d_in[3];
  const float* b2 = (const float*)d_in[4];
  const float* W3 = (const float*)d_in[5];
  const float* b3 = (const float*)d_in[6];
  const float* Ws = (const float*)d_in[7];
  const float* bs = (const float*)d_in[8];
  float* out = (float*)d_out;

  // Workspace: 4 bf16 activation buffers of Bsz*Hd (2 MB each, 8 MB total)
  const size_t actElems = (size_t)Bsz * Hd;
  __bf16* hping = (__bf16*)d_ws;
  __bf16* hpong = hping + actElems;
  __bf16* h2buf = hpong + actElems;
  __bf16* h3buf = h2buf + actElems;
  __bf16* hio[2] = {hping, hpong};

  // h0 = 0 for the first recurrence step (re-init every call: deterministic)
  (void)hipMemsetAsync(hping, 0, actElems * sizeof(__bf16), stream);

  dim3 grid(Bsz / MT, Hd / NT);  // 16 x 8 = 128 workgroups
  dim3 blk(256);
  dim3 lgrid(Bsz / 8), lblk(256);

  for (int l = 0; l < Lv; ++l) {
    const __bf16* hin = hio[l & 1];
    __bf16* h1 = hio[(l + 1) & 1];
    const int* sig = (l > 0) ? (sigma_list + (l - 1)) : nullptr;
    // layer 1 (sequential recurrence step, sigma one-hot fused in epilogue)
    fused_gemm_elu_bf16<<<grid, blk, 0, stream>>>(
        hin, W1 + (size_t)l * Hd * FDim, b1 + (size_t)l * Hd, sig, Lv, h1);
    // layer 2 (sigma = 0)
    fused_gemm_elu_bf16<<<grid, blk, 0, stream>>>(
        h1, W2 + (size_t)l * Hd * FDim, b2 + (size_t)l * Hd, nullptr, 0, h2buf);
    // layer 3 (sigma = 0)
    fused_gemm_elu_bf16<<<grid, blk, 0, stream>>>(
        h2buf, W3 + (size_t)l * Hd * FDim, b3 + (size_t)l * Hd, nullptr, 0, h3buf);
    // head: logits -> softmax[:,1]
    logits_sigmoid<<<lgrid, lblk, 0, stream>>>(
        h3buf, Ws + (size_t)l * 2 * Hd, bs + (size_t)l * 2, out, l);
  }
}
